// CrysFormer_12841952215474
// MI455X (gfx1250) — matrix operand order repaired and checked by reading, not compiled, not run on hardware
//
#include <hip/hip_runtime.h>
#include <hip/hip_bf16.h>
#include <math.h>

#define B_    64
#define NA_   24
#define N_    (B_*NA_)        // 1536
#define HID_  128
#define HEADS_ 8
#define NL_   4
#define LAT_  128
#define NF_   10
#define ED_   66
#define EDP_  96              // padded edge-feature width (K multiple of 32)
#define PDIM_ 80              // padded p width (N multiple of 16)
#define RDIM_ 96              // padded r width (K multiple of 32)
#define QKV_  (HEADS_*HID_)   // 1024
#define H3_   (3*HID_)        // 384

typedef __bf16 bf16_t;
typedef __attribute__((ext_vector_type(16))) __bf16 bf16x16;
typedef __attribute__((ext_vector_type(8)))  __bf16 bf16x8;
typedef __attribute__((ext_vector_type(8)))  float  floatx8;

__device__ __forceinline__ float warp_sum(float v) {
#pragma unroll
  for (int off = 16; off > 0; off >>= 1) v += __shfl_xor(v, off, 32);
  return v;
}

// ---------------------------------------------------------------------------
// Batched bf16 WMMA GEMM: C[z] = act(A[z] @ W[z] + bias + Cadd[z])
// W is PRE-PACKED into WMMA B-fragment order (see k_packB): per 32x16 K-N
// block, lane L's 16 contiguous bf16 are K = kb*32 + (L/16)*8 + (e%8) + (e/8)*16 at column
// nt*16 + (L%16). One wave computes a 32x16 C tile: two independent
// accumulators -> two back-to-back v_wmma with no RAW hazard.
// A is row-major; a lane's A fragment is two contiguous 8-elem runs.
// Requires M%32==0, Npad%16==0, K%32==0 (caller zero-pads).
// FLAGS: 1=bias, 2=Cadd, 4=SiLU, 8=store f32 C, 16=store bf16 Cb
// ---------------------------------------------------------------------------
template <int FLAGS>
__global__ __launch_bounds__(256) void k_gemm(
    const bf16_t* __restrict__ A, int lda, long aB,
    const bf16_t* __restrict__ W, long wB,
    const float* __restrict__ bias,
    const float* __restrict__ Cadd, long caB, int ldca,
    float* __restrict__ C, long cB, int ldc,
    bf16_t* __restrict__ Cb, long cbB, int ldcb,
    int M, int Npad, int K) {
  const int lane = threadIdx.x & 31;
  const int wv   = threadIdx.x >> 5;
  const int ntn  = Npad >> 4;
  const int ntm2 = M >> 5;
  const int tile = blockIdx.x * 8 + wv;
  if (tile >= ntm2 * ntn) return;
  const int mt = tile / ntn, nt = tile % ntn;
  const long zb = blockIdx.y;
  const int m = lane & 15;
  const int g = lane >> 4;
  const bf16_t* a0p = A + zb * aB + (long)(mt * 32 + m) * lda + 8 * g;
  const bf16_t* a1p = a0p + (long)16 * lda;
  const bf16_t* wp  = W + zb * wB + ((long)nt * 32 + lane) * 16;
  const long wstep  = (long)ntn * 512;   // 32 lanes * 16 elems per K-block row
  floatx8 acc0 = {0.f, 0.f, 0.f, 0.f, 0.f, 0.f, 0.f, 0.f};
  floatx8 acc1 = {0.f, 0.f, 0.f, 0.f, 0.f, 0.f, 0.f, 0.f};
  for (int k0 = 0; k0 < K; k0 += 32) {
    bf16x16 bv = *(const bf16x16*)wp;
    wp += wstep;
    bf16x8 a0lo = *(const bf16x8*)(a0p + k0);
    bf16x8 a0hi = *(const bf16x8*)(a0p + k0 + 16);
    bf16x8 a1lo = *(const bf16x8*)(a1p + k0);
    bf16x8 a1hi = *(const bf16x8*)(a1p + k0 + 16);
    bf16x16 av0 = __builtin_shufflevector(a0lo, a0hi, 0, 1, 2, 3, 4, 5, 6, 7,
                                          8, 9, 10, 11, 12, 13, 14, 15);
    bf16x16 av1 = __builtin_shufflevector(a1lo, a1hi, 0, 1, 2, 3, 4, 5, 6, 7,
                                          8, 9, 10, 11, 12, 13, 14, 15);
    acc0 = __builtin_amdgcn_wmma_f32_16x16x32_bf16(false, av0, false, bv,
                                                   (short)0, acc0, false, false);
    acc1 = __builtin_amdgcn_wmma_f32_16x16x32_bf16(false, av1, false, bv,
                                                   (short)0, acc1, false, false);
  }
  const int nn = nt * 16 + m;
  const float bval = (FLAGS & 1) ? bias[nn] : 0.f;
#pragma unroll
  for (int r = 0; r < 8; ++r) {
    const int mm0 = mt * 32 + r + 8 * g;   // ISA C/D layout: M = r + 8*(lane/16)
    const int mm1 = mm0 + 16;
    float v0 = acc0[r] + bval;
    float v1 = acc1[r] + bval;
    if (FLAGS & 2) {
      v0 += Cadd[zb * caB + (long)mm0 * ldca + nn];
      v1 += Cadd[zb * caB + (long)mm1 * ldca + nn];
    }
    if (FLAGS & 4) {
      v0 = v0 / (1.f + __expf(-v0));
      v1 = v1 / (1.f + __expf(-v1));
    }
    if (FLAGS & 8) {
      C[zb * cB + (long)mm0 * ldc + nn] = v0;
      C[zb * cB + (long)mm1 * ldc + nn] = v1;
    }
    if (FLAGS & 16) {
      Cb[zb * cbB + (long)mm0 * ldcb + nn] = (bf16_t)v0;
      Cb[zb * cbB + (long)mm1 * ldcb + nn] = (bf16_t)v1;
    }
  }
}

// Pack an f32 source matrix into bf16 WMMA B-fragment order with zero-padding.
// Element (z, k, n) is read from src[base + z*zsrc + k*sk + n*sn] when
// k<Kvalid && n<Nvalid, else 0.  dst layout per z: [(kb*ntn+nt)*32+lane]*16+e
// with k = kb*32 + (lane/16)*8 + (e%8) + (e/8)*16,  n = nt*16 + (lane%16).
__global__ void k_packB(const float* __restrict__ src, long base, long sk, long sn,
                        long zsrc, int Kvalid, int Nvalid, int K, int Npad,
                        bf16_t* __restrict__ dst, int total) {
  int idx = blockIdx.x * 256 + threadIdx.x;
  if (idx >= total) return;
  const int per = K * Npad;
  const int z = idx / per;
  const int rem = idx - z * per;
  const int e = rem & 15;
  const int lane = (rem >> 4) & 31;
  const int t = rem >> 9;
  const int ntn = Npad >> 4;
  const int kb = t / ntn, nt = t % ntn;
  const int k = kb * 32 + (lane >> 4) * 8 + (e & 7) + (e >> 3) * 16;
  const int n = nt * 16 + (lane & 15);
  float v = 0.f;
  if (k < Kvalid && n < Nvalid)
    v = src[base + (long)z * zsrc + (long)k * sk + (long)n * sn];
  dst[idx] = (bf16_t)v;
}

// node input: A0[row, 0:128] = type_emb[at[row]-1], A0[row,128:256] = t[b]
__global__ void k_node_in(const int* __restrict__ at, const float* __restrict__ t,
                          const float* __restrict__ temb, bf16_t* __restrict__ A0) {
  int idx = blockIdx.x * 256 + threadIdx.x;
  if (idx >= N_ * 256) return;
  int row = idx >> 8, c = idx & 255;
  float v = (c < 128) ? temb[(long)(at[row] - 1) * HID_ + c]
                      : t[(row / NA_) * LAT_ + (c - 128)];
  A0[idx] = (bf16_t)v;
}

// edge features + LayerNorm -> bf16 [B*NA*NA, EDP_]; one wave per edge
__global__ void k_edges(const float* __restrict__ frac, const float* __restrict__ latrep,
                        const float* __restrict__ eg, const float* __restrict__ eb,
                        bf16_t* __restrict__ ef) {
  const int e = blockIdx.x * 8 + (threadIdx.x >> 5);
  const int lane = threadIdx.x & 31;
  if (e >= B_ * NA_ * NA_) return;
  const int j = e % NA_, i = (e / NA_) % NA_, b = e / (NA_ * NA_);
  float fd[3];
#pragma unroll
  for (int a = 0; a < 3; ++a) {
    float d = frac[(b * NA_ + j) * 3 + a] - frac[(b * NA_ + i) * 3 + a];
    fd[a] = d - floorf(d);
  }
  const float TWO_PI = 6.28318530717958647692f;
  float vals[3]; float s = 0.f;
#pragma unroll
  for (int tt = 0; tt < 3; ++tt) {
    int c = lane + 32 * tt; float val = 0.f;
    if (c < 30)       { int a = c / NF_, f = c % NF_;        val = __sinf(fd[a] * TWO_PI * (float)f); }
    else if (c < 60)  { int cc = c - 30; int a = cc / NF_;   val = __cosf(fd[a] * TWO_PI * (float)(cc % NF_)); }
    else if (c < 66)  { val = latrep[b * 6 + (c - 60)]; }
    vals[tt] = val;
    if (c < ED_) s += val;
  }
  float mean = warp_sum(s) * (1.f / ED_);
  float vs = 0.f;
#pragma unroll
  for (int tt = 0; tt < 3; ++tt) {
    int c = lane + 32 * tt;
    if (c < ED_) { float d = vals[tt] - mean; vs += d * d; }
  }
  float rstd = rsqrtf(warp_sum(vs) * (1.f / ED_) + 1e-5f);
  bf16_t* o = ef + (long)e * EDP_;
#pragma unroll
  for (int tt = 0; tt < 3; ++tt) {
    int c = lane + 32 * tt;
    if (c < EDP_)
      o[c] = (c < ED_) ? (bf16_t)((vals[tt] - mean) * rstd * eg[c] + eb[c]) : (bf16_t)0.f;
  }
}

// LayerNorm over 128 features; one wave per row; writes f32 + bf16
__global__ void k_ln(const float* __restrict__ x, const float* __restrict__ g,
                     const float* __restrict__ b, float* __restrict__ y,
                     bf16_t* __restrict__ yb, int rows) {
  const int row = blockIdx.x * 8 + (threadIdx.x >> 5);
  const int lane = threadIdx.x & 31;
  if (row >= rows) return;
  const float* xp = x + (long)row * HID_;
  float v[4]; float s = 0.f;
#pragma unroll
  for (int tt = 0; tt < 4; ++tt) { v[tt] = xp[lane * 4 + tt]; s += v[tt]; }
  float mean = warp_sum(s) * (1.f / HID_);
  float vs = 0.f;
#pragma unroll
  for (int tt = 0; tt < 4; ++tt) { float d = v[tt] - mean; vs += d * d; }
  float rstd = rsqrtf(warp_sum(vs) * (1.f / HID_) + 1e-5f);
#pragma unroll
  for (int tt = 0; tt < 4; ++tt) {
    int c = lane * 4 + tt;
    float o = (v[tt] - mean) * rstd * g[c] + b[c];
    y[(long)row * HID_ + c] = o;
    yb[(long)row * HID_ + c] = (bf16_t)o;
  }
}

// attention: one wave per (b, j, h). logits include edge term via p; emits
// out_partial = sum_i a*v  and  r[c] = sum_i a*ef[b,i,j,c]  (bf16, zero-padded)
__global__ void k_attn(const float* __restrict__ q, const float* __restrict__ k,
                       const float* __restrict__ v, const float* __restrict__ p,
                       const bf16_t* __restrict__ ef,
                       float* __restrict__ outp, bf16_t* __restrict__ r) {
  const int wid = blockIdx.x * 8 + (threadIdx.x >> 5);
  const int lane = threadIdx.x & 31;
  if (wid >= B_ * NA_ * HEADS_) return;
  const int hh = wid & (HEADS_ - 1);
  const int j  = (wid >> 3) % NA_;
  const int b  = wid / (HEADS_ * NA_);
  const int rowj = b * NA_ + j;
  const float scale = 0.08838834764831845f;  // 1/sqrt(128)
  float qd[4];
  const float* qp = q + (long)rowj * QKV_ + hh * HID_;
#pragma unroll
  for (int tt = 0; tt < 4; ++tt) qd[tt] = qp[lane * 4 + tt];
  float pc[3];
#pragma unroll
  for (int tt = 0; tt < 3; ++tt) {
    int c = lane + 32 * tt;
    pc[tt] = (c < ED_) ? p[((long)rowj * HEADS_ + hh) * PDIM_ + c] : 0.f;
  }
  float logits[NA_];
  for (int i = 0; i < NA_; ++i) {
    const int rowi = b * NA_ + i;
    const float* kp = k + (long)rowi * QKV_ + hh * HID_;
    float s = 0.f;
#pragma unroll
    for (int tt = 0; tt < 4; ++tt) s += qd[tt] * kp[lane * 4 + tt];
    const bf16_t* efp = ef + ((long)(b * NA_ + i) * NA_ + j) * EDP_;
#pragma unroll
    for (int tt = 0; tt < 3; ++tt) {
      int c = lane + 32 * tt;
      if (c < ED_) s += (float)efp[c] * pc[tt];
    }
    logits[i] = warp_sum(s) * scale;
  }
  float mx = logits[0];
  for (int i = 1; i < NA_; ++i) mx = fmaxf(mx, logits[i]);
  float sum = 0.f;
  for (int i = 0; i < NA_; ++i) { logits[i] = __expf(logits[i] - mx); sum += logits[i]; }
  const float inv = 1.f / sum;
  float od[4] = {0.f, 0.f, 0.f, 0.f};
  float rc[3] = {0.f, 0.f, 0.f};
  for (int i = 0; i < NA_; ++i) {
    const float a = logits[i] * inv;
    const float* vp = v + (long)(b * NA_ + i) * QKV_ + hh * HID_;
#pragma unroll
    for (int tt = 0; tt < 4; ++tt) od[tt] += a * vp[lane * 4 + tt];
    const bf16_t* efp = ef + ((long)(b * NA_ + i) * NA_ + j) * EDP_;
#pragma unroll
    for (int tt = 0; tt < 3; ++tt) {
      int c = lane + 32 * tt;
      if (c < ED_) rc[tt] += a * (float)efp[c];
    }
  }
  float* op = outp + ((long)rowj * HEADS_ + hh) * HID_;
#pragma unroll
  for (int tt = 0; tt < 4; ++tt) op[lane * 4 + tt] = od[tt];
  bf16_t* rp = r + ((long)rowj * HEADS_ + hh) * RDIM_;
#pragma unroll
  for (int tt = 0; tt < 3; ++tt) {
    int c = lane + 32 * tt;
    if (c < RDIM_) rp[c] = (c < ED_) ? (bf16_t)rc[tt] : (bf16_t)0.f;
  }
}

// head mean + skip:  outm = mean_h(out2) + skip
__global__ void k_headmean(const float* __restrict__ out2, const float* __restrict__ skip,
                           float* __restrict__ outm) {
  int idx = blockIdx.x * 256 + threadIdx.x;
  if (idx >= N_ * HID_) return;
  int row = idx >> 7, d = idx & 127;
  float s = 0.f;
#pragma unroll
  for (int h = 0; h < HEADS_; ++h) s += out2[(long)row * QKV_ + h * HID_ + d];
  outm[idx] = s * (1.f / HEADS_) + skip[idx];
}

// x = bf16([u, v, u-v])
__global__ void k_concatx(const float* __restrict__ u, const float* __restrict__ v,
                          bf16_t* __restrict__ x) {
  int idx = blockIdx.x * 256 + threadIdx.x;
  if (idx >= N_ * HID_) return;
  int row = idx >> 7, d = idx & 127;
  float a = u[idx], bb = v[idx];
  bf16_t* xp = x + (long)row * H3_;
  xp[d] = (bf16_t)a; xp[HID_ + d] = (bf16_t)bb; xp[2 * HID_ + d] = (bf16_t)(a - bb);
}

// gate = sigmoid(g2 @ w3 + b3);  h = gate*u + (1-gate)*v ;  one wave per row
__global__ void k_gate(const bf16_t* __restrict__ g2, const float* __restrict__ w3,
                       const float* __restrict__ b3, const float* __restrict__ u,
                       const float* __restrict__ v, float* __restrict__ hout,
                       bf16_t* __restrict__ hb) {
  const int row = blockIdx.x * 8 + (threadIdx.x >> 5);
  const int lane = threadIdx.x & 31;
  if (row >= N_) return;
  float s = 0.f;
#pragma unroll
  for (int tt = 0; tt < 3; ++tt) {
    int c = lane + 32 * tt;  // covers 0..95 exactly
    s += (float)g2[(long)row * 96 + c] * w3[c];
  }
  float gate = 1.f / (1.f + __expf(-(warp_sum(s) + b3[0])));
#pragma unroll
  for (int tt = 0; tt < 4; ++tt) {
    int c = lane * 4 + tt;
    long idx = (long)row * HID_ + c;
    float o = gate * u[idx] + (1.f - gate) * v[idx];
    hout[idx] = o;
    hb[idx] = (bf16_t)o;
  }
}

// graph mean over atoms
__global__ void k_gfeat(const float* __restrict__ h, float* __restrict__ gf,
                        bf16_t* __restrict__ gfb) {
  int idx = blockIdx.x * 256 + threadIdx.x;
  if (idx >= B_ * HID_) return;
  int b = idx >> 7, d = idx & 127;
  float s = 0.f;
  for (int n = 0; n < NA_; ++n) s += h[(long)(b * NA_ + n) * HID_ + d];
  s *= (1.f / NA_);
  gf[idx] = s; gfb[idx] = (bf16_t)s;
}

__global__ void k_latfinal(const float* __restrict__ l2, const float* __restrict__ w,
                           float* __restrict__ out) {
  int idx = blockIdx.x * 256 + threadIdx.x;
  if (idx >= B_ * 6) return;
  int b = idx / 6, o = idx % 6;
  float s = 0.f;
  for (int c = 0; c < 32; ++c) s += l2[b * 32 + c] * w[c * 6 + o];
  out[idx] = s;
}

__global__ void k_frcfinal(const float* __restrict__ f2, const float* __restrict__ w,
                           float* __restrict__ out) {
  int idx = blockIdx.x * 256 + threadIdx.x;
  if (idx >= N_ * 3) return;
  int row = idx / 3, o = idx % 3;
  float s = 0.f;
  for (int c = 0; c < 32; ++c) s += f2[row * 32 + c] * w[c * 3 + o];
  out[idx] = s;
}

// ---------------------------------------------------------------------------
static void gemm(hipStream_t st, int flags,
                 const bf16_t* A, int lda, long aB,
                 const bf16_t* W, long wB,
                 const float* bias,
                 const float* Cadd, long caB, int ldca,
                 float* C, long cB, int ldc,
                 bf16_t* Cb, long cbB, int ldcb,
                 int M, int Npad, int K, int batch) {
  int tiles = (M / 32) * (Npad / 16);
  dim3 grid((tiles + 7) / 8, batch, 1);
#define GEMM_CASE(F)                                                          \
  case F:                                                                     \
    k_gemm<F><<<grid, 256, 0, st>>>(A, lda, aB, W, wB, bias, Cadd, caB, ldca, \
                                    C, cB, ldc, Cb, cbB, ldcb, M, Npad, K);   \
    break;
  switch (flags) {
    GEMM_CASE(8)
    GEMM_CASE(9)
    GEMM_CASE(10)
    GEMM_CASE(13)
    GEMM_CASE(21)
    GEMM_CASE(25)
    default: break;
  }
#undef GEMM_CASE
}

static void packB(hipStream_t st, const float* src, long base, long sk, long sn,
                  long zsrc, int Kvalid, int Nvalid, int K, int Npad,
                  bf16_t* dst, int nz) {
  int total = nz * K * Npad;
  k_packB<<<(total + 255) / 256, 256, 0, st>>>(src, base, sk, sn, zsrc, Kvalid,
                                               Nvalid, K, Npad, dst, total);
}

extern "C" void kernel_launch(void* const* d_in, const int* in_sizes, int n_in,
                              void* d_out, int out_size, void* d_ws, size_t ws_size,
                              hipStream_t stream) {
  (void)in_sizes; (void)n_in; (void)out_size; (void)ws_size;
  const float* t_in    = (const float*)d_in[0];
  const int*   at      = (const int*)d_in[1];
  const float* frac    = (const float*)d_in[2];
  const float* latrep  = (const float*)d_in[3];
  const float* temb    = (const float*)d_in[9];
  const float* align_w = (const float*)d_in[10];
  const float* align_b = (const float*)d_in[11];
  const float* eln_g   = (const float*)d_in[12];
  const float* eln_b   = (const float*)d_in[13];
  const float* ln1_g   = (const float*)d_in[14];
  const float* ln1_b   = (const float*)d_in[15];
  const float* qw      = (const float*)d_in[16];
  const float* qb      = (const float*)d_in[17];
  const float* kw      = (const float*)d_in[18];
  const float* kb      = (const float*)d_in[19];
  const float* vw      = (const float*)d_in[20];
  const float* vb      = (const float*)d_in[21];
  const float* ew      = (const float*)d_in[22];
  const float* skw     = (const float*)d_in[23];
  const float* skb     = (const float*)d_in[24];
  const float* ga1w    = (const float*)d_in[25];
  const float* ga1b    = (const float*)d_in[26];
  const float* ga2w    = (const float*)d_in[27];
  const float* ga2b    = (const float*)d_in[28];
  const float* ga3w    = (const float*)d_in[29];
  const float* ga3b    = (const float*)d_in[30];
  const float* ln2_g   = (const float*)d_in[31];
  const float* ln2_b   = (const float*)d_in[32];
  const float* ff1w    = (const float*)d_in[33];
  const float* ff1b    = (const float*)d_in[34];
  const float* ff2w    = (const float*)d_in[35];
  const float* ff2b    = (const float*)d_in[36];
  const float* gf1w    = (const float*)d_in[37];
  const float* gf1b    = (const float*)d_in[38];
  const float* gf2w    = (const float*)d_in[39];
  const float* gf2b    = (const float*)d_in[40];
  const float* gf3w    = (const float*)d_in[41];
  const float* gf3b    = (const float*)d_in[42];
  const float* lat1w   = (const float*)d_in[43];
  const float* lat1b   = (const float*)d_in[44];
  const float* lat2w   = (const float*)d_in[45];
  const float* lat2b   = (const float*)d_in[46];
  const float* lat3w   = (const float*)d_in[47];
  const float* frc1w   = (const float*)d_in[48];
  const float* frc1b   = (const float*)d_in[49];
  const float* frc2w   = (const float*)d_in[50];
  const float* frc2b   = (const float*)d_in[51];
  const float* frc3w   = (const float*)d_in[52];

  float* out_lat = (float*)d_out;
  float* out_frc = (float*)d_out + B_ * 6;

  size_t off = 0;
  auto alloc = [&](size_t bytes) -> void* {
    void* p = (char*)d_ws + off;
    off += (bytes + 255) & ~(size_t)255;
    return p;
  };
  bf16_t* efb  = (bf16_t*)alloc((size_t)B_ * NA_ * NA_ * EDP_ * 2);
  bf16_t* A0   = (bf16_t*)alloc((size_t)N_ * 256 * 2);
  float*  h    = (float*)alloc((size_t)N_ * HID_ * 4);
  bf16_t* hbf  = (bf16_t*)alloc((size_t)N_ * HID_ * 2);
  float*  hn   = (float*)alloc((size_t)N_ * HID_ * 4);
  bf16_t* hnb  = (bf16_t*)alloc((size_t)N_ * HID_ * 2);
  float*  hn2  = (float*)alloc((size_t)N_ * HID_ * 4);
  bf16_t* hn2b = (bf16_t*)alloc((size_t)N_ * HID_ * 2);
  float*  qf   = (float*)alloc((size_t)N_ * QKV_ * 4);
  bf16_t* qbf  = (bf16_t*)alloc((size_t)N_ * QKV_ * 2);
  float*  kf   = (float*)alloc((size_t)N_ * QKV_ * 4);
  float*  vf   = (float*)alloc((size_t)N_ * QKV_ * 4);
  float*  pbuf = (float*)alloc((size_t)N_ * HEADS_ * PDIM_ * 4);
  bf16_t* rbuf = (bf16_t*)alloc((size_t)N_ * HEADS_ * RDIM_ * 2);
  float*  out2 = (float*)alloc((size_t)N_ * QKV_ * 4);
  float*  skp  = (float*)alloc((size_t)N_ * HID_ * 4);
  float*  outm = (float*)alloc((size_t)N_ * HID_ * 4);
  bf16_t* xcat = (bf16_t*)alloc((size_t)N_ * H3_ * 2);
  bf16_t* gm1  = (bf16_t*)alloc((size_t)N_ * 192 * 2);
  bf16_t* gm2  = (bf16_t*)alloc((size_t)N_ * 96 * 2);
  bf16_t* ff1o = (bf16_t*)alloc((size_t)N_ * HID_ * 2);
  float*  ffo  = (float*)alloc((size_t)N_ * HID_ * 4);
  // bf16 fragment-packed weight staging
  bf16_t* wA   = (bf16_t*)alloc((size_t)256 * HID_ * 2);
  bf16_t* wq   = (bf16_t*)alloc((size_t)HID_ * QKV_ * 2);
  bf16_t* wk   = (bf16_t*)alloc((size_t)HID_ * QKV_ * 2);
  bf16_t* wv   = (bf16_t*)alloc((size_t)HID_ * QKV_ * 2);
  bf16_t* wsk  = (bf16_t*)alloc((size_t)HID_ * HID_ * 2);
  bf16_t* ewPh = (bf16_t*)alloc((size_t)HEADS_ * RDIM_ * HID_ * 2);
  bf16_t* ewT  = (bf16_t*)alloc((size_t)HEADS_ * HID_ * PDIM_ * 2);
  bf16_t* wg1  = (bf16_t*)alloc((size_t)H3_ * 192 * 2);
  bf16_t* wg2  = (bf16_t*)alloc((size_t)192 * 96 * 2);
  bf16_t* wf1  = (bf16_t*)alloc((size_t)HID_ * HID_ * 2);
  bf16_t* wf2  = (bf16_t*)alloc((size_t)HID_ * HID_ * 2);
  bf16_t* wl1  = (bf16_t*)alloc((size_t)HID_ * 64 * 2);
  bf16_t* wl2  = (bf16_t*)alloc((size_t)64 * 32 * 2);
  bf16_t* wr1  = (bf16_t*)alloc((size_t)HID_ * 64 * 2);
  bf16_t* wr2  = (bf16_t*)alloc((size_t)64 * 32 * 2);
  float*  gft  = (float*)alloc((size_t)B_ * HID_ * 4);
  bf16_t* gftb = (bf16_t*)alloc((size_t)B_ * HID_ * 2);
  bf16_t* l1o  = (bf16_t*)alloc((size_t)B_ * 64 * 2);
  float*  l2o  = (float*)alloc((size_t)B_ * 32 * 4);
  bf16_t* f1o  = (bf16_t*)alloc((size_t)N_ * 64 * 2);
  float*  f2o  = (float*)alloc((size_t)N_ * 32 * 4);

  // ---- static weight packing (row-major [K,N] -> B-fragment order) ----
  packB(stream, align_w, 0, 128, 1, 0, 256, 128, 256, 128, wA, 1);
  packB(stream, lat1w, 0, 64, 1, 0, 128, 64, 128, 64, wl1, 1);
  packB(stream, lat2w, 0, 32, 1, 0, 64, 32, 64, 32, wl2, 1);
  packB(stream, frc1w, 0, 64, 1, 0, 128, 64, 128, 64, wr1, 1);
  packB(stream, frc2w, 0, 32, 1, 0, 64, 32, 64, 32, wr2, 1);

  // ---- node & edge features ----
  k_node_in<<<(N_ * 256 + 255) / 256, 256, 0, stream>>>(at, t_in, temb, A0);
  k_edges<<<(B_ * NA_ * NA_ + 7) / 8, 256, 0, stream>>>(frac, latrep, eln_g, eln_b, efb);
  gemm(stream, 9, A0, 256, 0, wA, 0, align_b, nullptr, 0, 0,
       h, 0, HID_, nullptr, 0, 0, N_, HID_, 256, 1);

  for (int l = 0; l < NL_; ++l) {
    const float* ew_l = ew + (size_t)l * ED_ * QKV_;
    packB(stream, qw + (size_t)l * HID_ * QKV_, 0, QKV_, 1, 0, HID_, QKV_, HID_, QKV_, wq, 1);
    packB(stream, kw + (size_t)l * HID_ * QKV_, 0, QKV_, 1, 0, HID_, QKV_, HID_, QKV_, wk, 1);
    packB(stream, vw + (size_t)l * HID_ * QKV_, 0, QKV_, 1, 0, HID_, QKV_, HID_, QKV_, wv, 1);
    packB(stream, skw + (size_t)l * HID_ * HID_, 0, HID_, 1, 0, HID_, HID_, HID_, HID_, wsk, 1);
    packB(stream, ga1w + (size_t)l * H3_ * 192, 0, 192, 1, 0, H3_, 192, H3_, 192, wg1, 1);
    packB(stream, ga2w + (size_t)l * 192 * 96, 0, 96, 1, 0, 192, 96, 192, 96, wg2, 1);
    packB(stream, ff1w + (size_t)l * HID_ * HID_, 0, HID_, 1, 0, HID_, HID_, HID_, HID_, wf1, 1);
    packB(stream, ff2w + (size_t)l * HID_ * HID_, 0, HID_, 1, 0, HID_, HID_, HID_, HID_, wf2, 1);
    // ewT[h]: (k=d, n=c) = ew[c*1024 + h*128 + d]   (K=128, N=66->80)
    packB(stream, ew_l, 0, 1, QKV_, HID_, HID_, ED_, HID_, PDIM_, ewT, HEADS_);
    // ewPh[h]: (k=c, n=d) = ew[c*1024 + h*128 + d]  (K=66->96, N=128)
    packB(stream, ew_l, 0, QKV_, 1, HID_, ED_, HID_, RDIM_, HID_, ewPh, HEADS_);

    k_ln<<<N_ / 8, 256, 0, stream>>>(h, ln1_g + l * HID_, ln1_b + l * HID_, hn, hnb, N_);

    gemm(stream, 25, hnb, HID_, 0, wq, 0, qb + (size_t)l * QKV_, nullptr, 0, 0,
         qf, 0, QKV_, qbf, 0, QKV_, N_, QKV_, HID_, 1);
    gemm(stream, 9, hnb, HID_, 0, wk, 0, kb + (size_t)l * QKV_, nullptr, 0, 0,
         kf, 0, QKV_, nullptr, 0, 0, N_, QKV_, HID_, 1);
    gemm(stream, 9, hnb, HID_, 0, wv, 0, vb + (size_t)l * QKV_, nullptr, 0, 0,
         vf, 0, QKV_, nullptr, 0, 0, N_, QKV_, HID_, 1);
    gemm(stream, 9, hnb, HID_, 0, wsk, 0, skb + (size_t)l * HID_, nullptr, 0, 0,
         skp, 0, HID_, nullptr, 0, 0, N_, HID_, HID_, 1);

    // p[row,h,:] = q_h @ ew_h^T  (batched over heads)
    gemm(stream, 8, qbf, QKV_, HID_, ewT, (long)HID_ * PDIM_, nullptr,
         nullptr, 0, 0, pbuf, PDIM_, HEADS_ * PDIM_, nullptr, 0, 0,
         N_, PDIM_, HID_, HEADS_);

    k_attn<<<(B_ * NA_ * HEADS_) / 8, 256, 0, stream>>>(qf, kf, vf, pbuf, efb, out2, rbuf);

    // out2 += r @ ew_h  (batched over heads, in-place add)
    gemm(stream, 10, rbuf, HEADS_ * RDIM_, RDIM_, ewPh, (long)RDIM_ * HID_, nullptr,
         out2, HID_, QKV_, out2, HID_, QKV_, nullptr, 0, 0,
         N_, HID_, RDIM_, HEADS_);

    k_headmean<<<(N_ * HID_ + 255) / 256, 256, 0, stream>>>(out2, skp, outm);

    // gated residual 1: u=hn, v=outm
    k_concatx<<<(N_ * HID_ + 255) / 256, 256, 0, stream>>>(hn, outm, xcat);
    gemm(stream, 21, xcat, H3_, 0, wg1, 0, ga1b + (size_t)l * 192, nullptr, 0, 0,
         nullptr, 0, 0, gm1, 0, 192, N_, 192, H3_, 1);
    gemm(stream, 21, gm1, 192, 0, wg2, 0, ga2b + (size_t)l * 96, nullptr, 0, 0,
         nullptr, 0, 0, gm2, 0, 96, N_, 96, 192, 1);
    k_gate<<<N_ / 8, 256, 0, stream>>>(gm2, ga3w + (size_t)l * 96, ga3b + l,
                                       hn, outm, h, hbf);

    // FF block
    k_ln<<<N_ / 8, 256, 0, stream>>>(h, ln2_g + l * HID_, ln2_b + l * HID_, hn2, hn2b, N_);
    gemm(stream, 21, hn2b, HID_, 0, wf1, 0, ff1b + (size_t)l * HID_, nullptr, 0, 0,
         nullptr, 0, 0, ff1o, 0, HID_, N_, HID_, HID_, 1);
    gemm(stream, 9, ff1o, HID_, 0, wf2, 0, ff2b + (size_t)l * HID_, nullptr, 0, 0,
         ffo, 0, HID_, nullptr, 0, 0, N_, HID_, HID_, 1);

    // gated residual 2: u=hn2, v=ffo
    packB(stream, gf1w + (size_t)l * H3_ * 192, 0, 192, 1, 0, H3_, 192, H3_, 192, wg1, 1);
    packB(stream, gf2w + (size_t)l * 192 * 96, 0, 96, 1, 0, 192, 96, 192, 96, wg2, 1);
    k_concatx<<<(N_ * HID_ + 255) / 256, 256, 0, stream>>>(hn2, ffo, xcat);
    gemm(stream, 21, xcat, H3_, 0, wg1, 0, gf1b + (size_t)l * 192, nullptr, 0, 0,
         nullptr, 0, 0, gm1, 0, 192, N_, 192, H3_, 1);
    gemm(stream, 21, gm1, 192, 0, wg2, 0, gf2b + (size_t)l * 96, nullptr, 0, 0,
         nullptr, 0, 0, gm2, 0, 96, N_, 96, 192, 1);
    k_gate<<<N_ / 8, 256, 0, stream>>>(gm2, gf3w + (size_t)l * 96, gf3b + l,
                                       hn2, ffo, h, hbf);
  }

  // ---- output heads ----
  k_gfeat<<<(B_ * HID_ + 255) / 256, 256, 0, stream>>>(h, gft, gftb);
  gemm(stream, 21, gftb, HID_, 0, wl1, 0, lat1b, nullptr, 0, 0,
       nullptr, 0, 0, l1o, 0, 64, B_, 64, HID_, 1);
  gemm(stream, 13, l1o, 64, 0, wl2, 0, lat2b, nullptr, 0, 0,
       l2o, 0, 32, nullptr, 0, 0, B_, 32, 64, 1);
  k_latfinal<<<(B_ * 6 + 255) / 256, 256, 0, stream>>>(l2o, lat3w, out_lat);

  gemm(stream, 21, hbf, HID_, 0, wr1, 0, frc1b, nullptr, 0, 0,
       nullptr, 0, 0, f1o, 0, 64, N_, 64, HID_, 1);
  gemm(stream, 13, f1o, 64, 0, wr2, 0, frc2b, nullptr, 0, 0,
       f2o, 0, 32, nullptr, 0, 0, N_, 32, 64, 1);
  k_frcfinal<<<(N_ * 3 + 255) / 256, 256, 0, stream>>>(f2o, frc3w, out_frc);
}
